// PhiMoEDecoderLayer_89481348645455
// MI455X (gfx1250) — matrix operand order repaired and checked by reading, not compiled, not run on hardware
//
#include <hip/hip_runtime.h>

// ---------------------------------------------------------------------------
// PhiMoE decoder layer for MI455X (gfx1250): bf16 WMMA GEMMs, flash attention,
// sparse top-2 MoE with gather/scatter GEMMs, async LDS staging where dense.
// ---------------------------------------------------------------------------

typedef __bf16 bf16_t;
typedef __bf16 v16bf __attribute__((ext_vector_type(16)));
typedef float  v8f   __attribute__((ext_vector_type(8)));
typedef int    i32x4 __attribute__((vector_size(4 * sizeof(int))));

#define DEV __device__ __forceinline__

#if __has_builtin(__builtin_amdgcn_global_load_async_to_lds_b128)
#define HAVE_ASYNC_LDS 1
#endif

// 16-byte global->LDS copy: async (ASYNCcnt) when available, VGPR bounce else.
DEV void copy16_g2l(const bf16_t* g, bf16_t* l) {
#ifdef HAVE_ASYNC_LDS
  __builtin_amdgcn_global_load_async_to_lds_b128(
      (__attribute__((address_space(1))) i32x4*)g,
      (__attribute__((address_space(3))) i32x4*)l, 0, 0);
#else
  *(uint4*)l = *(const uint4*)g;
#endif
}
DEV void copy16_wait() {
#ifdef HAVE_ASYNC_LDS
#if __has_builtin(__builtin_amdgcn_s_wait_asynccnt)
  __builtin_amdgcn_s_wait_asynccnt(0);
#else
  asm volatile("s_wait_asynccnt 0" ::: "memory");
#endif
#endif
}

// ---- WMMA fragment loaders (CDNA5 ISA 7.12.2 layouts) ----------------------
// A (16x32 bf16): lane&15 = M row; half-wave interleaves K by 8 within each
// 16-wide K group.  elements 0..7 -> k0+half*8 .. +7 ; 8..15 -> k0+16+half*8.
DEV v16bf frag_ld_a(const bf16_t* tile, int ld, int rowBase, int k0) {
  int lane = threadIdx.x & 31;
  int row  = rowBase + (lane & 15);
  int half = lane >> 4;
  const bf16_t* p = tile + (size_t)row * ld + k0 + half * 8;
  union { uint4 u[2]; v16bf v; } f;
  f.u[0] = *(const uint4*)p;
  f.u[1] = *(const uint4*)(p + 16);   // +32B -> k0+16+half*8
  return f.v;
}
// B (32x16 bf16), stored transposed in LDS as [n][k]: lane&15 = N column;
// same K interleave as A: elements 0..7 -> k0+half*8, 8..15 -> k0+16+half*8.
DEV v16bf frag_ld_b(const bf16_t* tile, int ld, int rowBase, int k0) {
  int lane = threadIdx.x & 31;
  int row  = rowBase + (lane & 15);
  int half = lane >> 4;
  const bf16_t* p = tile + (size_t)row * ld + k0 + half * 8;
  union { uint4 u[2]; v16bf v; } f;
  f.u[0] = *(const uint4*)p;
  f.u[1] = *(const uint4*)(p + 16);   // +32B -> k0+16+half*8
  return f.v;
}

DEV v8f wmma_bf16(v16bf a, v16bf b, v8f c) {
  return __builtin_amdgcn_wmma_f32_16x16x32_bf16(
      /*neg_a=*/false, a, /*neg_b=*/false, b,
      /*c_mod=*/(short)0, c, /*reuse_a=*/false, /*reuse_b=*/false);
}

// ---------------------------------------------------------------------------
// Generic bf16 WMMA GEMM:  C[M,N] = gather(A)[M,K] @ B[K,N] (+bias)
//  - optional a_rows: A row gather indices
//  - optional c_rows/c_scale: scatter-add  C[c_rows[m]] += c_scale[m]*acc
//  - optional m_count: device-side effective M (ragged expert batches)
// Block: 256 threads (8 waves).  Tile: 128(M) x 64(N) x 32(K).
// ---------------------------------------------------------------------------
__global__ __launch_bounds__(256) void gemm_kernel(
    const bf16_t* __restrict__ A, int lda,
    const bf16_t* __restrict__ Bm, int ldb,
    float* __restrict__ C, int ldc,
    const float* __restrict__ bias,
    int M, int N, int K,
    const int*   __restrict__ a_rows,
    const int*   __restrict__ c_rows,
    const float* __restrict__ c_scale,
    const int*   __restrict__ m_count)
{
  __shared__ __align__(16) bf16_t As[128 * 40];
  __shared__ __align__(16) bf16_t Bs[64 * 40];

  int Meff = m_count ? *m_count : M;
  int mBase = blockIdx.y * 128;
  if (mBase >= Meff) return;
  int n0 = blockIdx.x * 64;
  int wave = threadIdx.x >> 5;
  int lane = threadIdx.x & 31;
  int lane16 = lane & 15, halfl = lane >> 4;

  bool denseA = (a_rows == nullptr) && (mBase + 128 <= Meff);

  v8f acc[4] = {};

  for (int k0 = 0; k0 < K; k0 += 32) {
    __syncthreads();
    // Stage A tile 128x32 (async when dense; zero-pad ragged rows otherwise)
    if (denseA) {
      for (int c = threadIdx.x; c < 512; c += 256) {
        int row = c >> 2, seg = (c & 3) * 8;
        copy16_g2l(A + (size_t)(mBase + row) * lda + k0 + seg,
                   &As[row * 40 + seg]);
      }
    } else {
      for (int c = threadIdx.x; c < 512; c += 256) {
        int row = c >> 2, seg = (c & 3) * 8;
        int m = mBase + row;
        uint4 val = make_uint4(0u, 0u, 0u, 0u);
        if (m < Meff) {
          int ar = a_rows ? a_rows[m] : m;
          val = *(const uint4*)(A + (size_t)ar * lda + k0 + seg);
        }
        *(uint4*)&As[row * 40 + seg] = val;
      }
    }
    // Stage B tile 32x64 transposed -> Bs[n][k]; prefetch next K-step tile
    for (int c = threadIdx.x; c < 256; c += 256) {
      int kk = c >> 3, seg = (c & 7) * 8;
      const bf16_t* src = Bm + (size_t)(k0 + kk) * ldb + n0 + seg;
      if (k0 + 32 < K)
        __builtin_prefetch(Bm + (size_t)(k0 + 32 + kk) * ldb + n0 + seg, 0, 1);
#pragma unroll
      for (int i = 0; i < 8; i++) Bs[(seg + i) * 40 + kk] = src[i];
    }
    copy16_wait();
    __syncthreads();

    // Load all fragments first, then back-to-back WMMAs.
    v16bf af  = frag_ld_a(As, 40, wave * 16, 0);
    v16bf bf0 = frag_ld_b(Bs, 40, 0,  0);
    v16bf bf1 = frag_ld_b(Bs, 40, 16, 0);
    v16bf bf2 = frag_ld_b(Bs, 40, 32, 0);
    v16bf bf3 = frag_ld_b(Bs, 40, 48, 0);
    acc[0] = wmma_bf16(af, bf0, acc[0]);
    acc[1] = wmma_bf16(af, bf1, acc[1]);
    acc[2] = wmma_bf16(af, bf2, acc[2]);
    acc[3] = wmma_bf16(af, bf3, acc[3]);
  }

  // Store C fragments: VGPR r -> row (half*8+r), col n0+t*16+lane16
#pragma unroll
  for (int t = 0; t < 4; t++) {
    int col = n0 + t * 16 + lane16;
#pragma unroll
    for (int r = 0; r < 8; r++) {
      int m = mBase + wave * 16 + halfl * 8 + r;
      if (m < Meff) {
        if (c_rows) {
          int orow = c_rows[m];
          float s = c_scale ? c_scale[m] : 1.0f;
          C[(size_t)orow * ldc + col] += s * acc[t][r];
        } else {
          float v = acc[t][r];
          if (bias) v += bias[col];
          C[(size_t)m * ldc + col] = v;
        }
      }
    }
  }
}

// ---------------------------------------------------------------------------
// Flash attention, causal, GQA (H=16 query heads, KV=4).
// Grid: (L/64, B*H).  Block: 128 threads = 4 waves, each wave = 16 query rows.
// ---------------------------------------------------------------------------
__global__ __launch_bounds__(128) void attn_kernel(
    const bf16_t* __restrict__ Qr,  // (B,H,L,64)
    const bf16_t* __restrict__ Kr,  // (B,KV,L,64)
    const bf16_t* __restrict__ Vr,  // (B,KV,L,64)
    bf16_t* __restrict__ Obf,       // (B,L,H*64)
    int Bn, int Hn, int KVn, int Ln)
{
  __shared__ __align__(16) bf16_t Ks[32 * 64];   // [key][hd]  == Bt for S
  __shared__ __align__(16) bf16_t Vs[64 * 32];   // [hd][key]  == Bt for PV
  __shared__ __align__(16) bf16_t Ps[4][16 * 32];

  int bh = blockIdx.y;
  int b = bh / Hn, h = bh % Hn;
  int kv = h / (Hn / KVn);
  int qbase = blockIdx.x * 64;
  int wave = threadIdx.x >> 5;
  int lane = threadIdx.x & 31;
  int lane16 = lane & 15, half = lane >> 4;

  const bf16_t* qp = Qr + (((size_t)(b * Hn + h)) * Ln + qbase + wave * 16) * 64;
  v16bf qf0 = frag_ld_a(qp, 64, 0, 0);
  v16bf qf1 = frag_ld_a(qp, 64, 0, 32);

  v8f o[4] = {};
  float mrow[8], lrow[8];
#pragma unroll
  for (int r = 0; r < 8; r++) { mrow[r] = -3e30f; lrow[r] = 0.f; }

  const bf16_t* kp = Kr + ((size_t)(b * KVn + kv)) * Ln * 64;
  const bf16_t* vp = Vr + ((size_t)(b * KVn + kv)) * Ln * 64;
  int nblocks = (qbase + 64 + 31) / 32;   // only key blocks touching causal band

  for (int j = 0; j < nblocks; j++) {
    __syncthreads();
    // Stage 32 keys: K async (row-major == Bt layout), V transposed via VGPRs
    for (int c = threadIdx.x; c < 256; c += 128) {
      int key = c >> 3, seg = (c & 7) * 8;
      copy16_g2l(kp + (size_t)(j * 32 + key) * 64 + seg, &Ks[key * 64 + seg]);
      const bf16_t* vsrc = vp + (size_t)(j * 32 + key) * 64 + seg;
#pragma unroll
      for (int i = 0; i < 8; i++) Vs[(seg + i) * 32 + key] = vsrc[i];
    }
    copy16_wait();
    __syncthreads();

    // S tiles (16 rows x 32 keys per wave)
    v8f s0 = {}, s1 = {};
    {
      v16bf b00 = frag_ld_b(Ks, 64, 0, 0);
      v16bf b01 = frag_ld_b(Ks, 64, 0, 32);
      v16bf b10 = frag_ld_b(Ks, 64, 16, 0);
      v16bf b11 = frag_ld_b(Ks, 64, 16, 32);
      s0 = wmma_bf16(qf0, b00, s0);
      s0 = wmma_bf16(qf1, b01, s0);
      s1 = wmma_bf16(qf0, b10, s1);
      s1 = wmma_bf16(qf1, b11, s1);
    }

    bf16_t* Pw = Ps[wave];
#pragma unroll
    for (int r = 0; r < 8; r++) {
      int qrow = qbase + wave * 16 + half * 8 + r;
      int key0 = j * 32 + lane16, key1 = key0 + 16;
      float v0 = s0[r] * 0.125f;
      float v1 = s1[r] * 0.125f;
      if (key0 > qrow) v0 = -1e30f;
      if (key1 > qrow) v1 = -1e30f;
      float mx = fmaxf(v0, v1);
#pragma unroll
      for (int offm = 1; offm < 16; offm <<= 1) mx = fmaxf(mx, __shfl_xor(mx, offm, 32));
      float mnew = fmaxf(mrow[r], mx);
      float scale = __expf(mrow[r] - mnew);
      float p0 = __expf(v0 - mnew), p1 = __expf(v1 - mnew);
      float ps = p0 + p1;
#pragma unroll
      for (int offm = 1; offm < 16; offm <<= 1) ps += __shfl_xor(ps, offm, 32);
      lrow[r] = lrow[r] * scale + ps;
      mrow[r] = mnew;
      o[0][r] *= scale; o[1][r] *= scale; o[2][r] *= scale; o[3][r] *= scale;
      Pw[(half * 8 + r) * 32 + lane16]      = (bf16_t)p0;
      Pw[(half * 8 + r) * 32 + 16 + lane16] = (bf16_t)p1;
    }
    // P written by this wave, read back by this wave: drain LDS counter
    asm volatile("s_wait_dscnt 0" ::: "memory");

    v16bf pf = frag_ld_a(Pw, 32, 0, 0);
    v16bf vb0 = frag_ld_b(Vs, 32, 0,  0);
    v16bf vb1 = frag_ld_b(Vs, 32, 16, 0);
    v16bf vb2 = frag_ld_b(Vs, 32, 32, 0);
    v16bf vb3 = frag_ld_b(Vs, 32, 48, 0);
    o[0] = wmma_bf16(pf, vb0, o[0]);
    o[1] = wmma_bf16(pf, vb1, o[1]);
    o[2] = wmma_bf16(pf, vb2, o[2]);
    o[3] = wmma_bf16(pf, vb3, o[3]);
  }

#pragma unroll
  for (int t = 0; t < 4; t++) {
#pragma unroll
    for (int r = 0; r < 8; r++) {
      int qrow = qbase + wave * 16 + half * 8 + r;
      float val = o[t][r] / lrow[r];
      Obf[((size_t)(b * Ln + qrow)) * (Hn * 64) + h * 64 + t * 16 + lane16] = (bf16_t)val;
    }
  }
}

// ---------------------------------------------------------------------------
// Elementwise / reduction helper kernels
// ---------------------------------------------------------------------------
__global__ __launch_bounds__(256) void rmsnorm_kernel(
    const float* __restrict__ x, const float* __restrict__ w,
    const float* __restrict__ bia, bf16_t* __restrict__ out_bf,
    float* __restrict__ out_f, int D)
{
  int row = blockIdx.x;
  const float* xr = x + (size_t)row * D;
  float ss = 0.f;
  for (int i = threadIdx.x; i < D; i += 256) { float v = xr[i]; ss += v * v; }
#pragma unroll
  for (int offm = 16; offm; offm >>= 1) ss += __shfl_xor(ss, offm, 32);
  __shared__ float red[8];
  __shared__ float rtot;
  if ((threadIdx.x & 31) == 0) red[threadIdx.x >> 5] = ss;
  __syncthreads();
  if (threadIdx.x == 0) {
    float t = 0.f;
    for (int i = 0; i < 8; i++) t += red[i];
    rtot = rsqrtf(t / (float)D + 1e-5f);
  }
  __syncthreads();
  float r = rtot;
  for (int i = threadIdx.x; i < D; i += 256) {
    float v = w[i] * (xr[i] * r) + bia[i];
    out_bf[(size_t)row * D + i] = (bf16_t)v;
    if (out_f) out_f[(size_t)row * D + i] = v;
  }
}

__global__ void convert_bf16_kernel(const float* __restrict__ in,
                                    bf16_t* __restrict__ out, size_t n)
{
  size_t i = (size_t)blockIdx.x * 256 + threadIdx.x;
  if (i < n) out[i] = (bf16_t)in[i];
}

// (B,L,Hh*64) f32 -> rope -> (B,Hh,L,64) bf16
__global__ void rope_kernel(const float* __restrict__ in,
                            const float* __restrict__ cosb,
                            const float* __restrict__ sinb,
                            bf16_t* __restrict__ out,
                            int Bn, int Ln, int Hh)
{
  size_t idx = (size_t)blockIdx.x * 256 + threadIdx.x;
  size_t total = (size_t)Bn * Hh * Ln * 64;
  if (idx >= total) return;
  int hd = idx & 63; size_t t = idx >> 6;
  int l = t % Ln; t /= Ln;
  int h = t % Hh; int b = t / Hh;
  const float* row = in + ((size_t)(b * Ln + l)) * (Hh * 64) + h * 64;
  float v = row[hd];
  float rot = (hd & 1) ? row[hd >> 1] : -row[32 + (hd >> 1)];
  out[idx] = (bf16_t)(v * cosb[l * 64 + hd] + rot * sinb[l * 64 + hd]);
}

// (B,L,Hh*64) f32 -> (B,Hh,L,64) bf16 (no rope; for V)
__global__ void vperm_kernel(const float* __restrict__ in,
                             bf16_t* __restrict__ out, int Bn, int Ln, int Hh)
{
  size_t idx = (size_t)blockIdx.x * 256 + threadIdx.x;
  size_t total = (size_t)Bn * Hh * Ln * 64;
  if (idx >= total) return;
  int hd = idx & 63; size_t t = idx >> 6;
  int l = t % Ln; t /= Ln;
  int h = t % Hh; int b = t / Hh;
  out[idx] = (bf16_t)in[((size_t)(b * Ln + l)) * (Hh * 64) + h * 64 + hd];
}

// x1 = x + y ; also seed d_out with x1 (MoE scatter adds on top)
__global__ void add_residual_kernel(const float* __restrict__ x,
                                    const float* __restrict__ y,
                                    float* __restrict__ x1,
                                    float* __restrict__ xout, size_t n)
{
  size_t i = (size_t)blockIdx.x * 256 + threadIdx.x;
  if (i < n) { float v = x[i] + y[i]; x1[i] = v; xout[i] = v; }
}

__global__ void router_kernel(const float* __restrict__ h2,
                              const float* __restrict__ gw,
                              float* __restrict__ logits,
                              int Ntok, int D, int E)
{
  int i = blockIdx.x * 256 + threadIdx.x;
  if (i >= Ntok * E) return;
  int e = i % E, tkn = i / E;
  const float* hr = h2 + (size_t)tkn * D;
  float acc = 0.f;
  for (int d = 0; d < D; d++) acc += hr[d] * gw[d * E + e];
  logits[i] = acc;
}

__global__ void topk_kernel(const float* __restrict__ logits,
                            int* __restrict__ counts,
                            int* __restrict__ rows,
                            float* __restrict__ wts, int Ntok, int E)
{
  int tkn = blockIdx.x * 256 + threadIdx.x;
  if (tkn >= Ntok) return;
  const float* l = logits + (size_t)tkn * E;
  float b0 = -3e38f, b1 = -3e38f; int i0 = 0, i1 = 0;
  for (int e = 0; e < E; e++) {
    float v = l[e];
    if (v > b0)      { b1 = b0; i1 = i0; b0 = v; i0 = e; }
    else if (v > b1) { b1 = v; i1 = e; }
  }
  float w0 = 1.f / (1.f + __expf(b1 - b0));
  float w1 = 1.f - w0;
  int p0 = atomicAdd(&counts[i0], 1);
  rows[i0 * Ntok + p0] = tkn; wts[i0 * Ntok + p0] = w0;
  int p1 = atomicAdd(&counts[i1], 1);
  rows[i1 * Ntok + p1] = tkn; wts[i1 * Ntok + p1] = w1;
}

__global__ void silu_mul_kernel(const float* __restrict__ g1,
                                const float* __restrict__ g3,
                                bf16_t* __restrict__ he,
                                const int* __restrict__ count, int F)
{
  size_t i = (size_t)blockIdx.x * 256 + threadIdx.x;
  int cnt = *count;
  if (i >= (size_t)cnt * F) return;
  float a = g1[i];
  float s = a / (1.f + __expf(-a));
  he[i] = (bf16_t)(s * g3[i]);
}

__global__ void zero_i32_kernel(int* p, int n)
{
  int i = threadIdx.x;
  if (i < n) p[i] = 0;
}

// ---------------------------------------------------------------------------
extern "C" void kernel_launch(void* const* d_in, const int* in_sizes, int n_in,
                              void* d_out, int out_size, void* d_ws, size_t ws_size,
                              hipStream_t stream)
{
  const int Bn = 8, Ln = 1024, Dm = 1024, Hn = 16, KVn = 4, En = 16, Fm = 2048;
  const int Ntok = Bn * Ln;  // 8192

  const float* x    = (const float*)d_in[0];
  const float* cosb = (const float*)d_in[1];
  const float* sinb = (const float*)d_in[2];
  const float* ln1w = (const float*)d_in[3];
  const float* ln1b = (const float*)d_in[4];
  const float* ln2w = (const float*)d_in[5];
  const float* ln2b = (const float*)d_in[6];
  const float* Wq   = (const float*)d_in[7];
  const float* bq   = (const float*)d_in[8];
  const float* Wk   = (const float*)d_in[9];
  const float* bk   = (const float*)d_in[10];
  const float* Wv   = (const float*)d_in[11];
  const float* bv   = (const float*)d_in[12];
  const float* Wo   = (const float*)d_in[13];
  const float* bo   = (const float*)d_in[14];
  const float* gw   = (const float*)d_in[15];
  const float* W1   = (const float*)d_in[16];
  const float* W2   = (const float*)d_in[17];
  const float* W3   = (const float*)d_in[18];

  float* xout   = (float*)d_out;                       // (B,L,D)
  float* logits = xout + (size_t)Ntok * Dm;            // (Ntok,E)

  size_t off = 0;
  auto alloc = [&](size_t bytes) -> void* {
    void* p = (char*)d_ws + off;
    off += (bytes + 255) & ~(size_t)255;
    return p;
  };

  bf16_t* h1   = (bf16_t*)alloc((size_t)Ntok * Dm * 2);
  float*  qf   = (float*) alloc((size_t)Ntok * 1024 * 4);
  float*  kf   = (float*) alloc((size_t)Ntok * 256 * 4);
  float*  vf   = (float*) alloc((size_t)Ntok * 256 * 4);
  bf16_t* qr   = (bf16_t*)alloc((size_t)Ntok * 1024 * 2);
  bf16_t* kr   = (bf16_t*)alloc((size_t)Ntok * 256 * 2);
  bf16_t* vt   = (bf16_t*)alloc((size_t)Ntok * 256 * 2);
  bf16_t* obf  = (bf16_t*)alloc((size_t)Ntok * 1024 * 2);
  float*  owo  = (float*) alloc((size_t)Ntok * Dm * 4);
  float*  x1   = (float*) alloc((size_t)Ntok * Dm * 4);
  bf16_t* h2b  = (bf16_t*)alloc((size_t)Ntok * Dm * 2);
  float*  h2f  = (float*) alloc((size_t)Ntok * Dm * 4);
  bf16_t* wqb  = (bf16_t*)alloc((size_t)1024 * 1024 * 2);
  bf16_t* wkb  = (bf16_t*)alloc((size_t)1024 * 256 * 2);
  bf16_t* wvb  = (bf16_t*)alloc((size_t)1024 * 256 * 2);
  bf16_t* wob  = (bf16_t*)alloc((size_t)1024 * 1024 * 2);
  int*    counts = (int*)  alloc(En * 4);
  int*    rows   = (int*)  alloc((size_t)En * Ntok * 4);
  float*  wts    = (float*)alloc((size_t)En * Ntok * 4);
  bf16_t* w1e  = (bf16_t*)alloc((size_t)Dm * Fm * 2);
  bf16_t* w3e  = (bf16_t*)alloc((size_t)Dm * Fm * 2);
  bf16_t* w2e  = (bf16_t*)alloc((size_t)Fm * Dm * 2);
  float*  g1   = (float*) alloc((size_t)Ntok * Fm * 4);
  float*  g3   = (float*) alloc((size_t)Ntok * Fm * 4);
  bf16_t* he   = (bf16_t*)alloc((size_t)Ntok * Fm * 2);

  auto cvt = [&](const float* src, bf16_t* dst, size_t n) {
    convert_bf16_kernel<<<(unsigned)((n + 255) / 256), 256, 0, stream>>>(src, dst, n);
  };
  auto gemm = [&](const bf16_t* A, int lda, const bf16_t* Bm, int ldb,
                  float* C, int ldc, const float* bias, int M, int N, int K,
                  const int* a_rows, const int* c_rows, const float* c_scale,
                  const int* cnt) {
    dim3 g(N / 64, (M + 127) / 128);
    gemm_kernel<<<g, 256, 0, stream>>>(A, lda, Bm, ldb, C, ldc, bias,
                                       M, N, K, a_rows, c_rows, c_scale, cnt);
  };

  // 0) expert counters
  zero_i32_kernel<<<1, 32, 0, stream>>>(counts, En);

  // 1) weight conversions (attention path)
  cvt(Wq, wqb, (size_t)1024 * 1024);
  cvt(Wk, wkb, (size_t)1024 * 256);
  cvt(Wv, wvb, (size_t)1024 * 256);
  cvt(Wo, wob, (size_t)1024 * 1024);

  // 2) rmsnorm1 -> h1 (bf16)
  rmsnorm_kernel<<<Ntok, 256, 0, stream>>>(x, ln1w, ln1b, h1, (float*)nullptr, Dm);

  // 3) Q/K/V projections (WMMA)
  gemm(h1, Dm, wqb, 1024, qf, 1024, bq, Ntok, 1024, Dm, nullptr, nullptr, nullptr, nullptr);
  gemm(h1, Dm, wkb, 256,  kf, 256,  bk, Ntok, 256,  Dm, nullptr, nullptr, nullptr, nullptr);
  gemm(h1, Dm, wvb, 256,  vf, 256,  bv, Ntok, 256,  Dm, nullptr, nullptr, nullptr, nullptr);

  // 4) RoPE + head-major permute, V permute
  {
    size_t nq = (size_t)Bn * Hn * Ln * 64;
    rope_kernel<<<(unsigned)((nq + 255) / 256), 256, 0, stream>>>(qf, cosb, sinb, qr, Bn, Ln, Hn);
    size_t nk = (size_t)Bn * KVn * Ln * 64;
    rope_kernel<<<(unsigned)((nk + 255) / 256), 256, 0, stream>>>(kf, cosb, sinb, kr, Bn, Ln, KVn);
    vperm_kernel<<<(unsigned)((nk + 255) / 256), 256, 0, stream>>>(vf, vt, Bn, Ln, KVn);
  }

  // 5) flash attention (WMMA), writes o in (B,L,H*HD) bf16
  attn_kernel<<<dim3(Ln / 64, Bn * Hn), 128, 0, stream>>>(qr, kr, vt, obf, Bn, Hn, KVn, Ln);

  // 6) output projection + residual (also seeds d_out)
  gemm(obf, 1024, wob, 1024, owo, 1024, bo, Ntok, 1024, 1024, nullptr, nullptr, nullptr, nullptr);
  {
    size_t n = (size_t)Ntok * Dm;
    add_residual_kernel<<<(unsigned)((n + 255) / 256), 256, 0, stream>>>(x, owo, x1, xout, n);
  }

  // 7) rmsnorm2 -> h2 (bf16 + f32)
  rmsnorm_kernel<<<Ntok, 256, 0, stream>>>(x1, ln2w, ln2b, h2b, h2f, Dm);

  // 8) router logits (f32, straight into d_out) + top-2 routing lists
  router_kernel<<<(Ntok * En + 255) / 256, 256, 0, stream>>>(h2f, gw, logits, Ntok, Dm, En);
  topk_kernel<<<(Ntok + 255) / 256, 256, 0, stream>>>(logits, counts, rows, wts, Ntok, En);

  // 9) sparse MoE: per expert, gathered GEMMs + scatter-add into d_out
  for (int e = 0; e < En; e++) {
    const int* rows_e = rows + (size_t)e * Ntok;
    const float* wts_e = wts + (size_t)e * Ntok;
    const int* cnt_e = counts + e;
    cvt(W1 + (size_t)e * Dm * Fm, w1e, (size_t)Dm * Fm);
    cvt(W3 + (size_t)e * Dm * Fm, w3e, (size_t)Dm * Fm);
    cvt(W2 + (size_t)e * Fm * Dm, w2e, (size_t)Fm * Dm);
    gemm(h2b, Dm, w1e, Fm, g1, Fm, nullptr, Ntok, Fm, Dm, rows_e, nullptr, nullptr, cnt_e);
    gemm(h2b, Dm, w3e, Fm, g3, Fm, nullptr, Ntok, Fm, Dm, rows_e, nullptr, nullptr, cnt_e);
    {
      size_t n = (size_t)Ntok * Fm;
      silu_mul_kernel<<<(unsigned)((n + 255) / 256), 256, 0, stream>>>(g1, g3, he, cnt_e, Fm);
    }
    // scatter: d_out[token] += w * (He @ W2e)   (experts serialized on stream)
    gemm(he, Fm, w2e, Dm, xout, Dm, nullptr, Ntok, Dm, Fm, nullptr, rows_e, wts_e, cnt_e);
  }
}
